// DCN_sep_pre_multi_offset_flow_similarity_41085657154043
// MI455X (gfx1250) — compile-verified, matrix operand order checked by reading, not run on hardware
//
#include <hip/hip_runtime.h>
#include <hip/hip_bf16.h>
#include <math.h>

// Problem constants (from reference)
#define BATCH 2
#define CIN   64
#define COUT  64
#define H     192
#define W     192
#define HW    (H*W)
#define DG    8
#define CG    8
#define KK    9
#define MAXMAG 10.0f

// Conv-offset-mask GEMM: M = B*H*W, N = 216 (pad 224), K = KK*CIN = 576
// K ordering: kidx = tap*64 + c  (half-rows of 8 = 8 consecutive channels)
#define CONV_N   216
#define CONV_NP  224
#define CONV_K   576
#define NT_CONV  14      // 224/16 total n-tiles
#define NT_HALF  7       // n-tiles per wave (2 waves per workgroup)
#define KC_CONV  18      // 576/32
#define MT_CONV  2       // M-tiles (of 16 px) per wave

// Deformable GEMM: M = B*H*W, N = 64, K = DG*KK*CG = 576 (order g*72 + tap*8 + c)
#define MD_K     576
#define KC_MD    18
#define NT_MD    4       // 64/16
#define MT_MD    4       // M-tiles per wave

typedef __bf16 v16bf __attribute__((ext_vector_type(16)));
typedef float  v8f   __attribute__((ext_vector_type(8)));

union Frag16 { unsigned short u[16]; uint4 q[2]; v16bf v; };

__device__ __forceinline__ unsigned short f2bf(float f) {
  unsigned int b = __float_as_uint(f);
  unsigned int r = b + 0x7FFFu + ((b >> 16) & 1u);   // round-to-nearest-even
  return (unsigned short)(r >> 16);
}
__device__ __forceinline__ float bf2f(unsigned short u) {
  return __uint_as_float(((unsigned int)u) << 16);
}
__device__ __forceinline__ int clampi(int v, int lo, int hi) {
  return v < lo ? lo : (v > hi ? hi : v);
}
// WMMA 16-bit A/B fragment element -> K index within a 32-wide chunk
__device__ __forceinline__ int frag_kin(int e, int hh) {
  return (e < 8) ? (hh * 8 + e) : (16 + hh * 8 + (e - 8));
}
// Swizzled B fragment: block (kc*NT + nt) is 32 lanes x 32B, lane loads 2x b128
__device__ __forceinline__ void load_bfrag(Frag16& f, const unsigned short* base,
                                           int blk, int lane) {
  const uint4* bp = (const uint4*)base + ((size_t)blk * 32 + lane) * 2;
  f.q[0] = bp[0];
  f.q[1] = bp[1];
}

// ---------------------------------------------------------------------------
// Convert x0 -> bf16 channel-last per group [B,DG,HW,CG];
//         x1 -> bf16 channel-last [B,HW,CIN]
// ---------------------------------------------------------------------------
__global__ void k_convert_inputs(const float* __restrict__ x0,
                                 const float* __restrict__ x1,
                                 unsigned short* __restrict__ x0t,
                                 unsigned short* __restrict__ x1t) {
  int idx = blockIdx.x * blockDim.x + threadIdx.x;
  const int total = BATCH * CIN * HW;
  if (idx >= total) return;
  int b   = idx / (CIN * HW);
  int rem = idx % (CIN * HW);
  int cin = rem / HW;
  int yx  = rem % HW;
  int g = cin / CG, c = cin % CG;
  x0t[(((size_t)(b * DG + g)) * HW + yx) * CG + c] = f2bf(x0[idx]);
  x1t[((size_t)b * HW + yx) * CIN + cin]           = f2bf(x1[idx]);
}

// ---------------------------------------------------------------------------
// Build B matrices directly in WMMA-fragment-swizzled layout.
//   convBs: kidx = tap*64 + c ; n in [0,224) zero-padded past 216
//   mdBs  : kidx = g*72 + tap*8 + c ; n = output channel
// ---------------------------------------------------------------------------
__global__ void k_convert_weights(const float* __restrict__ com_w,
                                  const float* __restrict__ weight,
                                  unsigned short* __restrict__ convBs,
                                  unsigned short* __restrict__ mdBs) {
  int idx = blockIdx.x * blockDim.x + threadIdx.x;
  // idx -> (blk, lane, e) with blk = kc*NT + nt
  int e    = idx & 15;
  int lane = (idx >> 4) & 31;
  int blk  = idx >> 9;
  int hh   = lane >> 4;
  int ncol = lane & 15;
  int kin  = frag_kin(e, hh);

  if (blk < KC_CONV * NT_CONV) {
    int kc = blk / NT_CONV, nt = blk % NT_CONV;
    int k = kc * 32 + kin;
    int n = nt * 16 + ncol;
    unsigned short v = 0;
    if (n < CONV_N) {
      int t = k / CIN;            // tap
      int c = k % CIN;            // input channel
      v = f2bf(com_w[(size_t)(n * CIN + c) * KK + t]);
    }
    convBs[idx] = v;
  }
  if (blk < KC_MD * NT_MD) {
    int kc = blk / NT_MD, nt = blk % NT_MD;
    int k = kc * 32 + kin;
    int o = nt * 16 + ncol;
    int g  = k / (KK * CG);
    int rk = (k % (KK * CG)) / CG;
    int c  = k % CG;
    mdBs[(size_t)blk * 512 + lane * 16 + e] =
        f2bf(weight[(size_t)(o * CIN + g * CG + c) * KK + rk]);
  }
}

// ---------------------------------------------------------------------------
// A-fragment builders
// ---------------------------------------------------------------------------
__device__ __forceinline__ Frag16 build_a_conv(const unsigned short* __restrict__ x1t,
                                               int b, int y, int x, int hh, int kc) {
  Frag16 a;
#pragma unroll
  for (int run = 0; run < 2; ++run) {
    int base = kc * 32 + run * 16 + hh * 8;    // multiple of 8
    int t  = base / CIN;                        // fixed tap for the run
    int c0 = base % CIN;                        // first of 8 channels
    int yy = y + (t / 3) - 1;
    int xx = x + (t % 3) - 1;
    uint4 q = make_uint4(0u, 0u, 0u, 0u);
    if (yy >= 0 && yy < H && xx >= 0 && xx < W)
      q = *(const uint4*)(x1t + ((size_t)b * HW + yy * W + xx) * CIN + c0);
    a.q[run] = q;
  }
  return a;
}

__device__ __forceinline__ Frag16 build_a_md(const unsigned short* __restrict__ x0t,
                                             const float* __restrict__ off,
                                             const float* __restrict__ mask,
                                             int b, int y, int x, int yx,
                                             int hh, int kc) {
  Frag16 a;
#pragma unroll
  for (int run = 0; run < 2; ++run) {
    int base = kc * 32 + run * 16 + hh * 8;    // multiple of 8 => fixed (g, tap)
    int g  = base / (KK * CG);
    int rk = (base % (KK * CG)) / CG;
    int ki = rk / 3, kj = rk % 3;

    size_t gk = ((size_t)(b * DG + g)) * KK + rk;
    float2 d2 = *(const float2*)(off + (gk * HW + yx) * 2);  // (dy,dx)
    float m   = mask[gk * HW + yx];

    float py = (float)(y - 1 + ki) + d2.x;
    float px = (float)(x - 1 + kj) + d2.y;
    float y0f = floorf(py), x0f = floorf(px);
    float ly = py - y0f, lx = px - x0f;
    int y0 = (int)y0f, xq = (int)x0f;

    bool vy0 = (y0 >= 0) & (y0 < H);
    bool vy1 = (y0 + 1 >= 0) & (y0 + 1 < H);
    bool vx0 = (xq >= 0) & (xq < W);
    bool vx1 = (xq + 1 >= 0) & (xq + 1 < W);
    float w00 = (vy0 & vx0) ? (1.0f - ly) * (1.0f - lx) : 0.0f;
    float w01 = (vy0 & vx1) ? (1.0f - ly) * lx : 0.0f;
    float w10 = (vy1 & vx0) ? ly * (1.0f - lx) : 0.0f;
    float w11 = (vy1 & vx1) ? ly * lx : 0.0f;

    int yc0 = clampi(y0, 0, H - 1), yc1 = clampi(y0 + 1, 0, H - 1);
    int xc0 = clampi(xq, 0, W - 1), xc1 = clampi(xq + 1, 0, W - 1);

    const unsigned short* xb = x0t + ((size_t)(b * DG + g)) * HW * CG;
    uint4 q00 = *(const uint4*)(xb + (size_t)(yc0 * W + xc0) * CG);
    uint4 q01 = *(const uint4*)(xb + (size_t)(yc0 * W + xc1) * CG);
    uint4 q10 = *(const uint4*)(xb + (size_t)(yc1 * W + xc0) * CG);
    uint4 q11 = *(const uint4*)(xb + (size_t)(yc1 * W + xc1) * CG);

    const unsigned int* u00 = (const unsigned int*)&q00;
    const unsigned int* u01 = (const unsigned int*)&q01;
    const unsigned int* u10 = (const unsigned int*)&q10;
    const unsigned int* u11 = (const unsigned int*)&q11;

#pragma unroll
    for (int c = 0; c < CG; ++c) {
      int d = c >> 1, s = (c & 1) * 16;
      float v00 = bf2f((unsigned short)(u00[d] >> s));
      float v01 = bf2f((unsigned short)(u01[d] >> s));
      float v10 = bf2f((unsigned short)(u10[d] >> s));
      float v11 = bf2f((unsigned short)(u11[d] >> s));
      float val = w00 * v00 + w01 * v01 + w10 * v10 + w11 * v11;
      a.u[run * 8 + c] = f2bf(val * m);
    }
  }
  return a;
}

// ---------------------------------------------------------------------------
// Kernel 1: implicit-GEMM 3x3 conv (x1 * com_w) + fused tanh/sigmoid epilogue.
// 2 waves per workgroup, each owning half the N range (7 n-tiles) for the
// same 32 pixels: keeps B-traffic amortization (full B read once per group)
// while accumulators stay at 112 VGPRs/wave (no VGPR-MSB switching).
// ---------------------------------------------------------------------------
__global__ void __launch_bounds__(64)
k_conv_offset_mask(const unsigned short* __restrict__ x1t,
                   const unsigned short* __restrict__ convBs,
                   const float* __restrict__ com_b,
                   const float* __restrict__ pre_offset,
                   const float* __restrict__ pre_sim,
                   float* __restrict__ off,    // [B,DG,KK,HW,2] (dy,dx)
                   float* __restrict__ mask) { // [B,DG,KK,HW]
  __shared__ float sOut[CONV_NP * (16 * MT_CONV)];   // [ch][px 0..31]
  const int tid  = threadIdx.x;
  const int wv   = tid >> 5;          // wave id: n-half
  const int lane = tid & 31;
  const int hh   = lane >> 4;
  const int ncol = lane & 15;
  const int ntbase = wv * NT_HALF;
  const int pbase  = blockIdx.x * (16 * MT_CONV);

  // Per-M-tile pixel coordinates for this lane's A rows
  int pb[MT_CONV], py_[MT_CONV], px_[MT_CONV];
#pragma unroll
  for (int mt = 0; mt < MT_CONV; ++mt) {
    int p = pbase + mt * 16 + ncol;
    pb[mt]  = p / HW;
    int yx  = p % HW;
    py_[mt] = yx / W;
    px_[mt] = yx % W;
  }

  v8f acc[MT_CONV][NT_HALF];
#pragma unroll
  for (int mt = 0; mt < MT_CONV; ++mt)
#pragma unroll
    for (int nt = 0; nt < NT_HALF; ++nt)
#pragma unroll
      for (int j = 0; j < 8; ++j) acc[mt][nt][j] = 0.0f;

  Frag16 a[MT_CONV];
#pragma unroll
  for (int mt = 0; mt < MT_CONV; ++mt)
    a[mt] = build_a_conv(x1t, pb[mt], py_[mt], px_[mt], hh, 0);
  Frag16 bcur;
  load_bfrag(bcur, convBs, ntbase, lane);

  for (int kc = 0; kc < KC_CONV; ++kc) {
    Frag16 anext[MT_CONV];
    if (kc + 1 < KC_CONV)
#pragma unroll
      for (int mt = 0; mt < MT_CONV; ++mt)
        anext[mt] = build_a_conv(x1t, pb[mt], py_[mt], px_[mt], hh, kc + 1);
#pragma unroll
    for (int nt = 0; nt < NT_HALF; ++nt) {
      Frag16 bnext;
      int nkc = kc, nnt = nt + 1;
      if (nnt == NT_HALF) { nnt = 0; nkc = kc + 1; }
      if (nkc < KC_CONV)
        load_bfrag(bnext, convBs, nkc * NT_CONV + ntbase + nnt, lane);
#pragma unroll
      for (int mt = 0; mt < MT_CONV; ++mt)
        acc[mt][nt] = __builtin_amdgcn_wmma_f32_16x16x32_bf16(
            false, a[mt].v, false, bcur.v, (short)0, acc[mt][nt], false, false);
      bcur = bnext;
    }
#pragma unroll
    for (int mt = 0; mt < MT_CONV; ++mt) a[mt] = anext[mt];
  }

  // Stage D through LDS, channel-major so px runs are contiguous (b128 stores)
#pragma unroll
  for (int mt = 0; mt < MT_CONV; ++mt)
#pragma unroll
    for (int nt = 0; nt < NT_HALF; ++nt) {
      int ch = (ntbase + nt) * 16 + ncol;
      float* row = &sOut[ch * (16 * MT_CONV) + mt * 16 + 8 * hh];
      *(float4*)(row)     = make_float4(acc[mt][nt][0], acc[mt][nt][1],
                                        acc[mt][nt][2], acc[mt][nt][3]);
      *(float4*)(row + 4) = make_float4(acc[mt][nt][4], acc[mt][nt][5],
                                        acc[mt][nt][6], acc[mt][nt][7]);
    }
  __syncthreads();

  // ---- Fused epilogue (both waves, 64-thread stride) ----
  for (int i = tid; i < (16 * MT_CONV) * CONV_N; i += 64) {
    int rr = i / CONV_N;
    int ch = i % CONV_N;
    int pp  = pbase + rr;
    int b2  = pp / HW;
    int yx2 = pp % HW;
    float v = sOut[ch * (16 * MT_CONV) + rr] + com_b[ch];
    if (ch < 144) {
      // channel = dg*18 + kk*2 + comp ; comp0=dy gets pre_offset[...,1]
      int dg   = ch / 18;
      int rem  = ch % 18;
      int kk   = rem >> 1;
      int comp = rem & 1;
      float o   = MAXMAG * tanhf(v);
      float pre = pre_offset[(((size_t)(b2 * KK + kk)) * HW + yx2) * 2 + (comp ^ 1)];
      off[((((size_t)(b2 * DG + dg)) * KK + kk) * HW + yx2) * 2 + comp] = o + pre;
    } else {
      int c2 = ch - 144;
      int dg = c2 / KK;
      int kk = c2 % KK;
      float sim = pre_sim[((size_t)(b2 * DG * KK + dg * KK + kk)) * HW + yx2];
      mask[(((size_t)(b2 * DG + dg)) * KK + kk) * HW + yx2] =
          1.0f / (1.0f + expf(-v * sim));
    }
  }
}

// ---------------------------------------------------------------------------
// Kernel 2: deformable conv as implicit GEMM.  One wave per 64 pixels
// (4 M-tiles): each B fragment feeds 4 WMMAs (4x less B L2 traffic).
// Output written as contiguous b128 pixel runs.
// ---------------------------------------------------------------------------
__global__ void __launch_bounds__(32)
k_mdcn(const unsigned short* __restrict__ x0t,
       const unsigned short* __restrict__ mdBs,
       const float* __restrict__ off,    // [B,DG,KK,HW,2]
       const float* __restrict__ mask,   // [B,DG,KK,HW]
       const float* __restrict__ bias,
       float* __restrict__ out) {
  const int lane = threadIdx.x;
  const int hh   = lane >> 4;
  const int ncol = lane & 15;
  const int pbase = blockIdx.x * (16 * MT_MD);

  int pb[MT_MD], yx_[MT_MD], py_[MT_MD], px_[MT_MD];
#pragma unroll
  for (int mt = 0; mt < MT_MD; ++mt) {
    int p = pbase + mt * 16 + ncol;
    pb[mt]  = p / HW;
    yx_[mt] = p % HW;
    py_[mt] = yx_[mt] / W;
    px_[mt] = yx_[mt] % W;
  }

  v8f acc[MT_MD][NT_MD];
#pragma unroll
  for (int mt = 0; mt < MT_MD; ++mt)
#pragma unroll
    for (int nt = 0; nt < NT_MD; ++nt)
#pragma unroll
      for (int j = 0; j < 8; ++j) acc[mt][nt][j] = 0.0f;

  Frag16 a[MT_MD];
#pragma unroll
  for (int mt = 0; mt < MT_MD; ++mt)
    a[mt] = build_a_md(x0t, off, mask, pb[mt], py_[mt], px_[mt], yx_[mt], hh, 0);
  Frag16 bcur;
  load_bfrag(bcur, mdBs, 0, lane);

  for (int kc = 0; kc < KC_MD; ++kc) {
    Frag16 anext[MT_MD];
    if (kc + 1 < KC_MD)
#pragma unroll
      for (int mt = 0; mt < MT_MD; ++mt)
        anext[mt] = build_a_md(x0t, off, mask, pb[mt], py_[mt], px_[mt],
                               yx_[mt], hh, kc + 1);
#pragma unroll
    for (int nt = 0; nt < NT_MD; ++nt) {
      Frag16 bnext;
      int nblk = kc * NT_MD + nt + 1;
      if (nblk < KC_MD * NT_MD) load_bfrag(bnext, mdBs, nblk, lane);
#pragma unroll
      for (int mt = 0; mt < MT_MD; ++mt)
        acc[mt][nt] = __builtin_amdgcn_wmma_f32_16x16x32_bf16(
            false, a[mt].v, false, bcur.v, (short)0, acc[mt][nt], false, false);
      bcur = bnext;
    }
#pragma unroll
    for (int mt = 0; mt < MT_MD; ++mt) a[mt] = anext[mt];
  }

  // Output: pixels j=0..7 (+8*hh) are contiguous flat indices -> b128 stores
  const int tb = pbase / HW;               // batch (64 | HW, no crossing)
#pragma unroll
  for (int mt = 0; mt < MT_MD; ++mt) {
    const int yxbase = (pbase % HW) + mt * 16 + 8 * hh;
#pragma unroll
    for (int nt = 0; nt < NT_MD; ++nt) {
      int ch = nt * 16 + ncol;
      float bv = bias[ch];
      float* dst = out + ((size_t)(tb * COUT + ch)) * HW + yxbase;
      *(float4*)(dst)     = make_float4(acc[mt][nt][0] + bv, acc[mt][nt][1] + bv,
                                        acc[mt][nt][2] + bv, acc[mt][nt][3] + bv);
      *(float4*)(dst + 4) = make_float4(acc[mt][nt][4] + bv, acc[mt][nt][5] + bv,
                                        acc[mt][nt][6] + bv, acc[mt][nt][7] + bv);
    }
  }
}

// ---------------------------------------------------------------------------
extern "C" void kernel_launch(void* const* d_in, const int* in_sizes, int n_in,
                              void* d_out, int out_size, void* d_ws, size_t ws_size,
                              hipStream_t stream) {
  const float* x0         = (const float*)d_in[0];
  const float* x1         = (const float*)d_in[1];
  const float* pre_offset = (const float*)d_in[2];
  const float* pre_sim    = (const float*)d_in[3];
  const float* weight     = (const float*)d_in[4];
  const float* bias       = (const float*)d_in[5];
  const float* com_w      = (const float*)d_in[6];
  const float* com_b      = (const float*)d_in[7];
  float* out = (float*)d_out;

  // Workspace layout (256B aligned segments)
  char* ws = (char*)d_ws;
  size_t o = 0;
  auto take = [&](size_t bytes) {
    size_t cur = o;
    o += (bytes + 255) & ~(size_t)255;
    return cur;
  };
  unsigned short* x0t    = (unsigned short*)(ws + take(sizeof(unsigned short) * (size_t)BATCH * CIN * HW));
  unsigned short* x1t    = (unsigned short*)(ws + take(sizeof(unsigned short) * (size_t)BATCH * CIN * HW));
  unsigned short* convBs = (unsigned short*)(ws + take(sizeof(unsigned short) * KC_CONV * NT_CONV * 512));
  unsigned short* mdBs   = (unsigned short*)(ws + take(sizeof(unsigned short) * KC_MD * NT_MD * 512));
  float* off  = (float*)(ws + take(sizeof(float) * (size_t)BATCH * DG * KK * 2 * HW));
  float* mask = (float*)(ws + take(sizeof(float) * (size_t)BATCH * DG * KK * HW));

  const int nconv = BATCH * CIN * HW;
  k_convert_inputs<<<(nconv + 255) / 256, 256, 0, stream>>>(x0, x1, x0t, x1t);

  const int nw = KC_CONV * NT_CONV * 512;   // covers mdBs range too
  k_convert_weights<<<(nw + 255) / 256, 256, 0, stream>>>(com_w, weight, convBs, mdBs);

  const int conv_blocks = BATCH * HW / (16 * MT_CONV);  // 2304 groups of 2 waves
  k_conv_offset_mask<<<conv_blocks, 64, 0, stream>>>(x1t, convBs, com_b,
                                                     pre_offset, pre_sim, off, mask);

  const int md_blocks = BATCH * HW / (16 * MT_MD);      // 1152
  k_mdcn<<<md_blocks, 32, 0, stream>>>(x0t, mdBs, off, mask, bias, out);
}
